// GCN_22694607192298
// MI455X (gfx1250) — compile-verified, matrix operand order checked by reading, not run on hardware
//
#include <hip/hip_runtime.h>
#include <hip/hip_bf16.h>

// ---------------------------------------------------------------------------
// GCN 3-layer forward for MI455X (gfx1250, wave32).
//   N=50000 nodes, E=800000 edges, dims 256 -> 256 -> 256 -> 128.
// GEMMs via v_wmma_f32_16x16x32_bf16 (bf16 in, f32 accum).
// Aggregation via global_atomic_add_f32 into an L2-resident buffer.
// ---------------------------------------------------------------------------

#define GCN_N      50000
#define GCN_E      800000
#define GCN_DIN    256
#define GCN_DHID   256
#define GCN_DOUT   128
#define GCN_EPS    1e-5f

typedef __attribute__((ext_vector_type(16))) __bf16 v16bf;
typedef __attribute__((ext_vector_type(8)))  float  v8f;

union FragBF {
    uint4 u[2];   // 32 bytes
    v16bf v;      // 16 x bf16
};

__device__ __forceinline__ unsigned short f32_to_bf16_rne(float f) {
    unsigned u = __float_as_uint(f);
    u += 0x7FFFu + ((u >> 16) & 1u);
    return (unsigned short)(u >> 16);
}

// -------------------------- small elementwise kernels -----------------------

__global__ void k_fill_f32(float* p, float v, int n) {
    int i = blockIdx.x * blockDim.x + threadIdx.x;
    if (i < n) p[i] = v;
}

__global__ void k_accum_deg(const int* __restrict__ row, float* __restrict__ deg, int e) {
    int i = blockIdx.x * blockDim.x + threadIdx.x;
    if (i < e) atomicAdd(&deg[row[i]], 1.0f);
}

__global__ void k_rsqrt_inplace(float* p, int n) {
    int i = blockIdx.x * blockDim.x + threadIdx.x;
    if (i < n) p[i] = rsqrtf(p[i]);
}

__global__ void k_f32_to_bf16(const float* __restrict__ in, unsigned short* __restrict__ out,
                              size_t n) {
    size_t i = (size_t)blockIdx.x * blockDim.x + threadIdx.x;
    if (i < n) out[i] = f32_to_bf16_rne(in[i]);
}

// W [K x C] row-major f32  ->  Wt [C x K] row-major bf16
__global__ void k_transpose_bf16(const float* __restrict__ W, unsigned short* __restrict__ Wt,
                                 int K, int C) {
    int i = blockIdx.x * blockDim.x + threadIdx.x;
    if (i < K * C) {
        int k = i / C, c = i % C;
        Wt[(size_t)c * K + k] = f32_to_bf16_rne(W[i]);
    }
}

// agg[i,c] = hw[i,c] * dsq[i]^2 + bias[c]   (self-loop + bias init)
__global__ void k_init_agg(const float* __restrict__ hw, const float* __restrict__ dsq,
                           const float* __restrict__ bias, float* __restrict__ agg,
                           int C, size_t total) {
    size_t i = (size_t)blockIdx.x * blockDim.x + threadIdx.x;
    if (i < total) {
        size_t node = i / (size_t)C;
        int   c     = (int)(i - node * C);
        float s     = dsq[node];
        agg[i] = hw[i] * s * s + bias[c];
    }
}

// ------------------------------ WMMA GEMM -----------------------------------
// C[m,n] = sum_k A[m,k] * W[k,n],  A bf16 [N x 256], Wt bf16 [Ccols x 256].
// Block = 4 waves; one 16-row tile in LDS shared by 4 waves, each wave owns a
// 16-wide column tile. 8 x (16x16x32) WMMA steps cover K=256.
__global__ void __launch_bounds__(128)
k_gemm_bf16_wmma(const unsigned short* __restrict__ Abf,
                 const unsigned short* __restrict__ Wt,
                 float* __restrict__ out, int Ccols) {
    __shared__ unsigned short As[16 * 256];   // 8 KB

    const int m0 = blockIdx.x * 16;

    // cooperative 16x256 bf16 tile load (512 x uint4)
    {
        const uint4* src = (const uint4*)(Abf + (size_t)m0 * 256);
        uint4*       dst = (uint4*)As;
        #pragma unroll
        for (int i = threadIdx.x; i < 512; i += 128) dst[i] = src[i];
    }
    __syncthreads();

    const int wave = threadIdx.x >> 5;
    const int lane = threadIdx.x & 31;
    const int r    = lane & 15;     // row (A) / col (B) within tile
    const int hi   = lane >> 4;     // K-half selector
    const int n0   = (blockIdx.y * 4 + wave) * 16;

    v8f acc = {};
    const unsigned short* brow = Wt + (size_t)(n0 + r) * 256;

    #pragma unroll
    for (int kt = 0; kt < 8; ++kt) {
        const int kk = kt * 32;
        FragBF a, b;
        // A fragment: lanes 0-15 -> K kk+0..7 & kk+16..23 ; lanes 16-31 -> +8
        const uint4* ap = (const uint4*)(As + r * 256 + kk + hi * 8);
        a.u[0] = ap[0];
        a.u[1] = ap[2];
        // B fragment: column n0+r of W == row of Wt, same K packing
        const uint4* bp = (const uint4*)(brow + kk + hi * 8);
        b.u[0] = bp[0];
        b.u[1] = bp[2];
        acc = __builtin_amdgcn_wmma_f32_16x16x32_bf16(
                  false, a.v, false, b.v, (short)0, acc, false, false);
    }

    // C/D layout: VGPR j -> row m0 + hi*8 + j, col n0 + (lane&15)
    float* o = out + (size_t)(m0 + hi * 8) * Ccols + (n0 + r);
    #pragma unroll
    for (int j = 0; j < 8; ++j) o[(size_t)j * Ccols] = acc[j];
}

// ----------------------------- edge scatter ---------------------------------
// one wave per edge: agg[dst, :] += hw[src, :] * (dsq[dst]*dsq[src])
__global__ void __launch_bounds__(256)
k_scatter_edges(const int* __restrict__ row, const int* __restrict__ col,
                const float* __restrict__ dsq, const float* __restrict__ hw,
                float* __restrict__ agg, int C, int E) {
    const int lane = threadIdx.x & 31;
    const int wid  = (blockIdx.x * blockDim.x + threadIdx.x) >> 5;
    const int nwv  = (gridDim.x * blockDim.x) >> 5;
    for (int e = wid; e < E; e += nwv) {
        const int dst = row[e];
        const int src = col[e];
        const float w = dsq[dst] * dsq[src];
        const float* hs = hw  + (size_t)src * C;
        float*       ad = agg + (size_t)dst * C;
        for (int j = lane; j < C; j += 32)
            atomicAdd(&ad[j], hs[j] * w);
    }
}

// ------------------------------- batch norm ---------------------------------

__global__ void k_zero_f32(float* p, int n) {
    int i = blockIdx.x * blockDim.x + threadIdx.x;
    if (i < n) p[i] = 0.0f;
}

// blockDim.x == C (256). Each block reduces `rows` rows, atomically folds in.
__global__ void __launch_bounds__(256)
k_bn_stats(const float* __restrict__ h, float* __restrict__ sums, int C, int n, int rows) {
    const int c  = threadIdx.x;
    const int r0 = blockIdx.x * rows;
    int r1 = r0 + rows; if (r1 > n) r1 = n;
    float s = 0.0f, s2 = 0.0f;
    for (int r = r0; r < r1; ++r) {
        float v = h[(size_t)r * C + c];
        s  += v;
        s2 += v * v;
    }
    atomicAdd(&sums[c], s);
    atomicAdd(&sums[C + c], s2);
}

// normalize + ReLU + convert to bf16 for the next layer's GEMM
__global__ void k_bn_apply_relu_bf16(const float* __restrict__ h, const float* __restrict__ sums,
                                     const float* __restrict__ gamma, const float* __restrict__ beta,
                                     unsigned short* __restrict__ out, int C, size_t total,
                                     float invN) {
    size_t i = (size_t)blockIdx.x * blockDim.x + threadIdx.x;
    if (i < total) {
        int c = (int)(i % (size_t)C);
        float mu  = sums[c] * invN;
        float var = sums[C + c] * invN - mu * mu;
        float v = (h[i] - mu) * rsqrtf(var + GCN_EPS) * gamma[c] + beta[c];
        v = fmaxf(v, 0.0f);
        out[i] = f32_to_bf16_rne(v);
    }
}

// ------------------------------ log-softmax ---------------------------------
// one wave per row of 128; each lane holds 4 values; shfl_xor reductions.
__global__ void __launch_bounds__(256)
k_log_softmax128(const float* __restrict__ h, float* __restrict__ out, int n) {
    const int lane = threadIdx.x & 31;
    const int wid  = (blockIdx.x * blockDim.x + threadIdx.x) >> 5;
    if (wid >= n) return;
    const float4 v = ((const float4*)(h + (size_t)wid * 128))[lane];
    float m = fmaxf(fmaxf(v.x, v.y), fmaxf(v.z, v.w));
    #pragma unroll
    for (int off = 16; off; off >>= 1) m = fmaxf(m, __shfl_xor(m, off, 32));
    float s = expf(v.x - m) + expf(v.y - m) + expf(v.z - m) + expf(v.w - m);
    #pragma unroll
    for (int off = 16; off; off >>= 1) s += __shfl_xor(s, off, 32);
    const float lg = m + logf(s);
    float4 o; o.x = v.x - lg; o.y = v.y - lg; o.z = v.z - lg; o.w = v.w - lg;
    ((float4*)(out + (size_t)wid * 128))[lane] = o;
}

// ------------------------------- launcher -----------------------------------

static inline size_t align256(size_t v) { return (v + 255) & ~(size_t)255; }

extern "C" void kernel_launch(void* const* d_in, const int* in_sizes, int n_in,
                              void* d_out, int out_size, void* d_ws, size_t ws_size,
                              hipStream_t stream) {
    (void)in_sizes; (void)n_in; (void)out_size; (void)ws_size;

    const float* x      = (const float*)d_in[0];
    const int*   eidx   = (const int*)d_in[1];          // [2 x E] int32
    const float* W0     = (const float*)d_in[2];
    const float* b0     = (const float*)d_in[3];
    const float* W1     = (const float*)d_in[4];
    const float* b1     = (const float*)d_in[5];
    const float* W2     = (const float*)d_in[6];
    const float* b2     = (const float*)d_in[7];
    const float* gamma0 = (const float*)d_in[8];
    const float* beta0  = (const float*)d_in[9];
    const float* gamma1 = (const float*)d_in[10];
    const float* beta1  = (const float*)d_in[11];
    float* outp = (float*)d_out;

    const int* erow = eidx;
    const int* ecol = eidx + GCN_E;

    // ---- workspace carve-up (~128.3 MB) ----
    char*  ws  = (char*)d_ws;
    size_t off = 0;
    float* dsq = (float*)(ws + off);                 off += align256((size_t)GCN_N * 4);
    unsigned short* hbf = (unsigned short*)(ws + off); off += align256((size_t)GCN_N * 256 * 2);
    float* hw  = (float*)(ws + off);                 off += align256((size_t)GCN_N * 256 * 4);
    float* agg = (float*)(ws + off);                 off += align256((size_t)GCN_N * 256 * 4);
    unsigned short* wt = (unsigned short*)(ws + off); off += align256((size_t)256 * 256 * 2);
    float* sums = (float*)(ws + off);                off += align256(512 * 4);

    const int TPB = 256;
    const size_t nfeat  = (size_t)GCN_N * 256;
    const size_t nfeat2 = (size_t)GCN_N * 128;

    // ---- degree (with self loop) -> 1/sqrt(deg) ----
    k_fill_f32<<<(GCN_N + TPB - 1) / TPB, TPB, 0, stream>>>(dsq, 1.0f, GCN_N);
    k_accum_deg<<<(GCN_E + TPB - 1) / TPB, TPB, 0, stream>>>(erow, dsq, GCN_E);
    k_rsqrt_inplace<<<(GCN_N + TPB - 1) / TPB, TPB, 0, stream>>>(dsq, GCN_N);

    // ---- x -> bf16 ----
    k_f32_to_bf16<<<(unsigned)((nfeat + TPB - 1) / TPB), TPB, 0, stream>>>(x, hbf, nfeat);

    const dim3 gemm_blk(128);
    const dim3 gemm_grid_256(GCN_N / 16, 256 / 64);
    const dim3 gemm_grid_128(GCN_N / 16, 128 / 64);
    const int  scat_blocks = 8192;
    const int  bn_rows     = 128;
    const int  bn_blocks   = (GCN_N + bn_rows - 1) / bn_rows;
    const float invN       = 1.0f / (float)GCN_N;

    // ============================ layer 0 ============================
    k_transpose_bf16<<<(256 * 256 + TPB - 1) / TPB, TPB, 0, stream>>>(W0, wt, 256, 256);
    k_gemm_bf16_wmma<<<gemm_grid_256, gemm_blk, 0, stream>>>(hbf, wt, hw, 256);
    k_init_agg<<<(unsigned)((nfeat + TPB - 1) / TPB), TPB, 0, stream>>>(hw, dsq, b0, agg, 256, nfeat);
    k_scatter_edges<<<scat_blocks, TPB, 0, stream>>>(erow, ecol, dsq, hw, agg, 256, GCN_E);
    k_zero_f32<<<2, 256, 0, stream>>>(sums, 512);
    k_bn_stats<<<bn_blocks, 256, 0, stream>>>(agg, sums, 256, GCN_N, bn_rows);
    k_bn_apply_relu_bf16<<<(unsigned)((nfeat + TPB - 1) / TPB), TPB, 0, stream>>>(
        agg, sums, gamma0, beta0, hbf, 256, nfeat, invN);

    // ============================ layer 1 ============================
    k_transpose_bf16<<<(256 * 256 + TPB - 1) / TPB, TPB, 0, stream>>>(W1, wt, 256, 256);
    k_gemm_bf16_wmma<<<gemm_grid_256, gemm_blk, 0, stream>>>(hbf, wt, hw, 256);
    k_init_agg<<<(unsigned)((nfeat + TPB - 1) / TPB), TPB, 0, stream>>>(hw, dsq, b1, agg, 256, nfeat);
    k_scatter_edges<<<scat_blocks, TPB, 0, stream>>>(erow, ecol, dsq, hw, agg, 256, GCN_E);
    k_zero_f32<<<2, 256, 0, stream>>>(sums, 512);
    k_bn_stats<<<bn_blocks, 256, 0, stream>>>(agg, sums, 256, GCN_N, bn_rows);
    k_bn_apply_relu_bf16<<<(unsigned)((nfeat + TPB - 1) / TPB), TPB, 0, stream>>>(
        agg, sums, gamma1, beta1, hbf, 256, nfeat, invN);

    // ============================ layer 2 ============================
    k_transpose_bf16<<<(256 * 128 + TPB - 1) / TPB, TPB, 0, stream>>>(W2, wt, 256, 128);
    k_gemm_bf16_wmma<<<gemm_grid_128, gemm_blk, 0, stream>>>(hbf, wt, hw, 128);
    k_init_agg<<<(unsigned)((nfeat2 + TPB - 1) / TPB), TPB, 0, stream>>>(hw, dsq, b2, agg, 128, nfeat2);
    k_scatter_edges<<<scat_blocks, TPB, 0, stream>>>(erow, ecol, dsq, hw, agg, 128, GCN_E);

    // ---- log-softmax (wave per row) ----
    const int ls_waves_per_blk = TPB / 32;
    const int ls_blocks = (GCN_N + ls_waves_per_blk - 1) / ls_waves_per_blk;
    k_log_softmax128<<<ls_blocks, TPB, 0, stream>>>(agg, outp, GCN_N);
}
